// ComfyNunchakuZImageFeedForward_50749333570204
// MI455X (gfx1250) — matrix operand order repaired and checked by reading, not compiled, not run on hardware
//
#include <hip/hip_runtime.h>
#include <hip/hip_bf16.h>
#include <stdint.h>

// SwiGLU FFN for MI455X (gfx1250, wave32, WMMA).
//   g  = clamp_fp16( silu(x @ w13[H:2H].T) * (x @ w13[0:H].T) )   [fused, kernel 1]
//   out = g @ w2.T                                                 [kernel 2]
// f16 WMMA (16x16x32) with f32 accumulation; g kept as f16 in d_ws (128 MB).
//
// L2 blocking strategy (192 MB L2, 23.3 TB/s HBM):
//   - blockIdx.x = M-band (fastest): activation tensor (x: 100 MB / g: 128 MB)
//     stays L2-resident across all N-columns; weights stream exactly once.
//   - weight loads are non-temporal so the streamed 201 MB / 100 MB weight
//     matrices do not evict the resident activations.
//   - final output stored non-temporal (never re-read).

typedef __attribute__((ext_vector_type(16))) _Float16 v16h;
typedef __attribute__((ext_vector_type(8)))  _Float16 v8h;
typedef __attribute__((ext_vector_type(2)))  __fp16   fp16x2;   // cvt_pkrtz result type
typedef __attribute__((ext_vector_type(8)))  float    v8f;
typedef __attribute__((ext_vector_type(4)))  float    v4f;

#define DIM   3072
#define HID   8192
#define BM    128
#define BN    64
#define BK    32
#define LSTR  (BK + 8)          // padded LDS row stride in halves (80B, 16B-aligned)
#define FP16_MAX_F 65504.0f

union AFrag { v16h v; v8h h[2]; };

__device__ __forceinline__ uint32_t pk2(float a, float b) {
    union { fp16x2 h; uint32_t u; } c;
    c.h = __builtin_amdgcn_cvt_pkrtz(a, b);   // v_cvt_pk_rtz_f16_f32
    return c.u;
}

// ---------------------------------------------------------------------------
// Kernel 1: fused [w1;w3] projection + SwiGLU, writes g (f16) to workspace.
// Grid: (M/BM, HID/BN)  — M fastest so x stays in L2, w13 streams once (NT).
// Block: 256 threads = 8 waves (4 x 2 wave grid).
// ---------------------------------------------------------------------------
__global__ __launch_bounds__(256)
void ffn_gemm1_swiglu(const float* __restrict__ x,
                      const float* __restrict__ w13,
                      _Float16* __restrict__ g) {
    __shared__ _Float16 Xs [BM][LSTR];
    __shared__ _Float16 W3s[BN][LSTR];
    __shared__ _Float16 W1s[BN][LSTR];

    const int tid  = threadIdx.x;
    const int lane = tid & 31;
    const int wave = tid >> 5;
    const int wm   = wave >> 1;   // 0..3 -> M offset wm*32
    const int wn   = wave & 1;    // 0..1 -> N offset wn*32
    const int lm   = lane & 15;   // row/col within 16
    const int lh   = lane >> 4;   // lane half

    const int mBase = blockIdx.x * BM;   // over tokens (fastest -> L2 reuse of x)
    const int nBase = blockIdx.y * BN;   // over HID

    v8f acc3[2][2] = {};   // x3 = h[:, n]        (first half of w13)
    v8f acc1[2][2] = {};   // x1 = h[:, H + n]    (second half of w13)

    for (int k0 = 0; k0 < DIM; k0 += BK) {
        // ---- stage X tile: BM x BK f32 -> f16 (1024 float4, 4 per thread) ----
        #pragma unroll
        for (int i = 0; i < 4; ++i) {
            int s  = tid + i * 256;
            int r  = s >> 3, c4 = s & 7;
            const float* src = &x[(mBase + r) * DIM + k0 + c4 * 4];
            if (i == 0 && k0 + BK < DIM) __builtin_prefetch(src + BK, 0, 1);
            v4f v = *(const v4f*)src;             // RT: keep x hot in L2
            *(uint2*)&Xs[r][c4 * 4] = make_uint2(pk2(v.x, v.y), pk2(v.z, v.w));
        }
        // ---- stage W3 (rows n) and W1 (rows H+n) tiles: 2 x (BN x BK), NT ----
        #pragma unroll
        for (int i = 0; i < 4; ++i) {
            int s  = tid + i * 256;          // 0..511 -> W3, 512..1023 -> W1
            int t  = s & 511;
            int r  = t >> 3, c4 = t & 7;
            int gr = (s < 512) ? (nBase + r) : (HID + nBase + r);
            v4f v = __builtin_nontemporal_load(
                        (const v4f*)&w13[gr * DIM + k0 + c4 * 4]);
            uint2 p = make_uint2(pk2(v.x, v.y), pk2(v.z, v.w));
            if (s < 512) *(uint2*)&W3s[r][c4 * 4] = p;
            else         *(uint2*)&W1s[r][c4 * 4] = p;
        }
        __syncthreads();

        // ---- fragments (ISA 7.12.2 wave32 16-bit layouts) ----
        AFrag a[2], b3[2], b1[2];
        #pragma unroll
        for (int sm = 0; sm < 2; ++sm) {
            int row = wm * 32 + sm * 16 + lm;
            a[sm].h[0] = *(const v8h*)&Xs[row][lh * 8];        // K = lh*8 .. +7
            a[sm].h[1] = *(const v8h*)&Xs[row][lh * 8 + 16];   // K = 16+lh*8 .. +7
        }
        #pragma unroll
        for (int sn = 0; sn < 2; ++sn) {
            int col = wn * 32 + sn * 16 + lm;
            b3[sn].h[0] = *(const v8h*)&W3s[col][lh * 8];      // K = lh*8 .. +7
            b3[sn].h[1] = *(const v8h*)&W3s[col][lh * 8 + 16]; // K = 16+lh*8 .. +7
            b1[sn].h[0] = *(const v8h*)&W1s[col][lh * 8];
            b1[sn].h[1] = *(const v8h*)&W1s[col][lh * 8 + 16];
        }
        #pragma unroll
        for (int sm = 0; sm < 2; ++sm)
            #pragma unroll
            for (int sn = 0; sn < 2; ++sn) {
                acc3[sm][sn] = __builtin_amdgcn_wmma_f32_16x16x32_f16(
                    false, a[sm].v, false, b3[sn].v, (short)0, acc3[sm][sn], false, false);
                acc1[sm][sn] = __builtin_amdgcn_wmma_f32_16x16x32_f16(
                    false, a[sm].v, false, b1[sn].v, (short)0, acc1[sm][sn], false, false);
            }
        __syncthreads();
    }

    // ---- SwiGLU epilogue in registers: g = clamp(silu(x1) * x3) -> f16 ----
    // Regular-temporal stores: g is re-read by kernel 2 and should land in L2.
    #pragma unroll
    for (int sm = 0; sm < 2; ++sm)
        #pragma unroll
        for (int sn = 0; sn < 2; ++sn) {
            int col = nBase + wn * 32 + sn * 16 + lm;
            #pragma unroll
            for (int r = 0; r < 8; ++r) {
                int row = mBase + wm * 32 + sm * 16 + lh * 8 + r;
                float x1v = acc1[sm][sn][r];
                float x3v = acc3[sm][sn][r];
                float sv  = x1v / (1.0f + __expf(-x1v));   // silu
                float gv  = sv * x3v;
                gv = fminf(fmaxf(gv, -FP16_MAX_F), FP16_MAX_F);
                g[(size_t)row * HID + col] = (_Float16)gv;
            }
        }
}

// ---------------------------------------------------------------------------
// Kernel 2: down projection  out[M, DIM] = g[M, HID](f16) @ w2[DIM, HID]^T.
// Grid: (M/BM, DIM/BN) — M fastest so g (128 MB) stays in L2, w2 streams (NT).
// Block: 256 threads = 8 waves.
// ---------------------------------------------------------------------------
__global__ __launch_bounds__(256)
void ffn_gemm2(const _Float16* __restrict__ g,
               const float* __restrict__ w2,
               float* __restrict__ out) {
    __shared__ _Float16 Gs[BM][LSTR];
    __shared__ _Float16 Ws[BN][LSTR];

    const int tid  = threadIdx.x;
    const int lane = tid & 31;
    const int wave = tid >> 5;
    const int wm   = wave >> 1;
    const int wn   = wave & 1;
    const int lm   = lane & 15;
    const int lh   = lane >> 4;

    const int mBase = blockIdx.x * BM;   // over tokens (fastest -> L2 reuse of g)
    const int nBase = blockIdx.y * BN;   // over DIM

    v8f acc[2][2] = {};

    for (int k0 = 0; k0 < HID; k0 += BK) {
        // ---- stage G tile: BM x BK halves, raw 16B copies (512 chunks) ----
        #pragma unroll
        for (int i = 0; i < 2; ++i) {
            int s = tid + i * 256;
            int r = s >> 2, c8 = s & 3;
            const _Float16* src = &g[(size_t)(mBase + r) * HID + k0 + c8 * 8];
            if (i == 0 && k0 + BK < HID) __builtin_prefetch(src + BK, 0, 1);
            *(uint4*)&Gs[r][c8 * 8] = *(const uint4*)src;   // RT: keep g hot in L2
        }
        // ---- stage W2 tile: BN x BK f32 -> f16 (512 float4), NT ----
        #pragma unroll
        for (int i = 0; i < 2; ++i) {
            int s = tid + i * 256;
            int r = s >> 3, c4 = s & 7;
            v4f v = __builtin_nontemporal_load(
                        (const v4f*)&w2[(size_t)(nBase + r) * HID + k0 + c4 * 4]);
            *(uint2*)&Ws[r][c4 * 4] = make_uint2(pk2(v.x, v.y), pk2(v.z, v.w));
        }
        __syncthreads();

        AFrag a[2], b[2];
        #pragma unroll
        for (int sm = 0; sm < 2; ++sm) {
            int row = wm * 32 + sm * 16 + lm;
            a[sm].h[0] = *(const v8h*)&Gs[row][lh * 8];
            a[sm].h[1] = *(const v8h*)&Gs[row][lh * 8 + 16];
        }
        #pragma unroll
        for (int sn = 0; sn < 2; ++sn) {
            int col = wn * 32 + sn * 16 + lm;
            b[sn].h[0] = *(const v8h*)&Ws[col][lh * 8];
            b[sn].h[1] = *(const v8h*)&Ws[col][lh * 8 + 16];
        }
        #pragma unroll
        for (int sm = 0; sm < 2; ++sm)
            #pragma unroll
            for (int sn = 0; sn < 2; ++sn)
                acc[sm][sn] = __builtin_amdgcn_wmma_f32_16x16x32_f16(
                    false, a[sm].v, false, b[sn].v, (short)0, acc[sm][sn], false, false);
        __syncthreads();
    }

    // out is never re-read: non-temporal stores, don't pollute L2.
    #pragma unroll
    for (int sm = 0; sm < 2; ++sm)
        #pragma unroll
        for (int sn = 0; sn < 2; ++sn) {
            int col = nBase + wn * 32 + sn * 16 + lm;
            #pragma unroll
            for (int r = 0; r < 8; ++r) {
                int row = mBase + wm * 32 + sm * 16 + lh * 8 + r;
                __builtin_nontemporal_store(acc[sm][sn][r],
                                            &out[(size_t)row * DIM + col]);
            }
        }
}

// ---------------------------------------------------------------------------
extern "C" void kernel_launch(void* const* d_in, const int* in_sizes, int n_in,
                              void* d_out, int out_size, void* d_ws, size_t ws_size,
                              hipStream_t stream) {
    const float* x   = (const float*)d_in[0];   // [B,S,DIM] f32
    const float* w13 = (const float*)d_in[1];   // [2*HID, DIM] f32
    const float* w2  = (const float*)d_in[2];   // [DIM, HID] f32
    float*    out = (float*)d_out;              // [B,S,DIM] f32
    _Float16* g   = (_Float16*)d_ws;            // [M, HID] f16 intermediate

    const int M = in_sizes[0] / DIM;            // B*S = 8192

    dim3 blk(256);
    dim3 grd1(M / BM, HID / BN);                // M fastest: x L2-resident
    ffn_gemm1_swiglu<<<grd1, blk, 0, stream>>>(x, w13, g);

    dim3 grd2(M / BM, DIM / BN);                // M fastest: g L2-resident
    ffn_gemm2<<<grd2, blk, 0, stream>>>(g, w2, out);
}
